// HMAEDecoder_65154653880508
// MI455X (gfx1250) — compile-verified, matrix operand order checked by reading, not run on hardware
//
#include <hip/hip_runtime.h>
#include <math.h>

// ---------------------------------------------------------------------------
// HMAE decoder for MI455X (gfx1250): all GEMMs via v_wmma_f32_16x16x32_bf16.
// Weights pre-transposed + converted to bf16 [N,K] once per launch so the
// weight path streams bf4 vectors (global) -> b64 stores (LDS) with no
// scalar scatter. Software-pipelined K loop hides global latency under WMMA.
// ---------------------------------------------------------------------------

typedef __attribute__((ext_vector_type(16))) __bf16 v16bf;
typedef __attribute__((ext_vector_type(8)))  float  v8f;
typedef __attribute__((ext_vector_type(4)))  __bf16 bf4;
typedef __attribute__((ext_vector_type(4)))  float  f4;

#define LDT 36  // padded LDS row stride (bf16 elems): 72B rows, 8B-aligned vec4 slots

template <typename T> struct Vec4;
template <> struct Vec4<float>  { typedef f4  type; };
template <> struct Vec4<__bf16> { typedef bf4 type; };

__device__ __forceinline__ __bf16 to_bf(float x)  { return (__bf16)x; }
__device__ __forceinline__ __bf16 to_bf(__bf16 x) { return x; }

// Load one 16x32 bf16 WMMA operand fragment from an LDS tile stored [row][k].
// CDNA5 16-bit A/B layout: lane L holds row (L&15); lane group (L>>4) selects
// K 0-7/16-23 vs 8-15/24-31. Assembled from 4 aligned b64 LDS loads.
__device__ __forceinline__ v16bf frag_ld(const __bf16* tile, int rowBase, int lane) {
    const __bf16* row = tile + (rowBase + (lane & 15)) * LDT + ((lane >> 4) << 3);
    bf4 p0 = *(const bf4*)(row);
    bf4 p1 = *(const bf4*)(row + 4);
    bf4 p2 = *(const bf4*)(row + 16);
    bf4 p3 = *(const bf4*)(row + 20);
    v16bf f;
#pragma unroll
    for (int j = 0; j < 4; ++j) {
        f[j]      = p0[j];
        f[4 + j]  = p1[j];
        f[8 + j]  = p2[j];
        f[12 + j] = p3[j];
    }
    return f;
}

// Generic batched WMMA GEMM:  C = act( scale * A @ B + bias [+ C] )
// A: [M,K] row-major (AT = float|__bf16), batch offset b1*sA0 + b2*sA1
// B: BTRANS ? [N,K] (addr n*ldb+k, BT = float|__bf16) : [K,N] (addr k*ldb+n)
// C: [M,N] fp32. grid.z = batch1*H2; 64x64 tile per block, 128 threads.
// REQUIRES: M % 64 == 0, K % 32 == 0.
template <typename AT, typename BT, bool BTRANS, bool GELU, bool ADDC>
__global__ __launch_bounds__(128) void gemm_wmma(
    const AT* __restrict__ A, long long sA0, long long sA1, int lda,
    const BT* __restrict__ Bm, long long sB0, long long sB1, int ldb,
    const float* __restrict__ bias,
    float* __restrict__ C, long long sC0, long long sC1, int ldc,
    int M, int N, int K, int H2, float scale)
{
    __shared__ __bf16 As[64 * LDT];
    __shared__ __bf16 Bs[64 * LDT];
    using AV4 = typename Vec4<AT>::type;
    using BV4 = typename Vec4<BT>::type;

    const int b1 = blockIdx.z / H2, b2 = blockIdx.z % H2;
    const AT* Ab = A  + (long long)b1 * sA0 + (long long)b2 * sA1;
    const BT* Bb = Bm + (long long)b1 * sB0 + (long long)b2 * sB1;
    float*    Cb = C  + (long long)b1 * sC0 + (long long)b2 * sC1;

    const int m0   = blockIdx.y * 64;
    const int n0   = blockIdx.x * 64;
    const int tid  = threadIdx.x;
    const int lane = tid & 31;
    const int w    = tid >> 5;
    const int wm   = (w >> 1) * 32;
    const int wn   = (w & 1) * 32;

    // staging geometry (vector-4 per thread, 4 iterations cover a 64x32 tile)
    const int amr = tid >> 3;           // A/BTRANS row 0..15 (+16 per iter)
    const int akc = (tid & 7) << 2;     // k chunk 0,4,...,28
    const int bkr = tid >> 4;           // !BTRANS: k row 0..7 (+8 per iter)
    const int bnc = (tid & 15) << 2;    // !BTRANS: n chunk 0,4,...,60

    const AT* aptr = Ab + (long long)(m0 + amr) * lda + akc;
    const long long aStep = (long long)16 * lda;

    AV4 ar[4];
    BV4 br[4];

    auto loadA = [&](int k0) {
#pragma unroll
        for (int i = 0; i < 4; ++i)
            ar[i] = *(const AV4*)(aptr + k0 + i * aStep);
    };
    auto loadB = [&](int k0) {
        if (BTRANS) {
#pragma unroll
            for (int i = 0; i < 4; ++i) {
                int r = n0 + amr + i * 16;
                r = r < N ? r : N - 1;      // clamp tail; junk cols never stored
                br[i] = *(const BV4*)(Bb + (long long)r * ldb + k0 + akc);
            }
        } else {
#pragma unroll
            for (int i = 0; i < 4; ++i) {
                long long krow = (long long)(k0 + bkr + i * 8) * ldb;
                if (n0 + bnc + 3 < N) {
                    br[i] = *(const BV4*)(Bb + krow + n0 + bnc);
                } else {
#pragma unroll
                    for (int j = 0; j < 4; ++j) {
                        int n = n0 + bnc + j;
                        n = n < N ? n : N - 1;
                        br[i][j] = Bb[krow + n];
                    }
                }
            }
        }
    };
    auto stageAB = [&]() {
#pragma unroll
        for (int i = 0; i < 4; ++i) {
            bf4 t;
#pragma unroll
            for (int j = 0; j < 4; ++j) t[j] = to_bf(ar[i][j]);
            *(bf4*)(As + (amr + i * 16) * LDT + akc) = t;
        }
        if (BTRANS) {
#pragma unroll
            for (int i = 0; i < 4; ++i) {
                bf4 t;
#pragma unroll
                for (int j = 0; j < 4; ++j) t[j] = to_bf(br[i][j]);
                *(bf4*)(Bs + (amr + i * 16) * LDT + akc) = t;
            }
        } else {
#pragma unroll
            for (int i = 0; i < 4; ++i) {
                int k = bkr + i * 8;
#pragma unroll
                for (int j = 0; j < 4; ++j)
                    Bs[(bnc + j) * LDT + k] = to_bf(br[i][j]);
            }
        }
    };

    v8f acc[2][2] = {};

    loadA(0);
    loadB(0);
    for (int k0 = 0; k0 < K; k0 += 32) {
        stageAB();
        __syncthreads();
        if (k0 + 32 < K) {            // prefetch next tile while WMMAs run
            loadA(k0 + 32);
            loadB(k0 + 32);
        }
        v16bf a0 = frag_ld(As, wm,      lane);
        v16bf a1 = frag_ld(As, wm + 16, lane);
        v16bf b0 = frag_ld(Bs, wn,      lane);
        v16bf b1f= frag_ld(Bs, wn + 16, lane);
        acc[0][0] = __builtin_amdgcn_wmma_f32_16x16x32_bf16(false, a0, false, b0,  (short)0, acc[0][0], false, false);
        acc[0][1] = __builtin_amdgcn_wmma_f32_16x16x32_bf16(false, a0, false, b1f, (short)0, acc[0][1], false, false);
        acc[1][0] = __builtin_amdgcn_wmma_f32_16x16x32_bf16(false, a1, false, b0,  (short)0, acc[1][0], false, false);
        acc[1][1] = __builtin_amdgcn_wmma_f32_16x16x32_bf16(false, a1, false, b1f, (short)0, acc[1][1], false, false);
        __syncthreads();
    }

    // epilogue: D layout -> lane holds col (lane&15); VGPR r is row r + 8*(lane>>4)
    const int ln = lane & 15, lg = lane >> 4;
#pragma unroll
    for (int i = 0; i < 2; ++i) {
#pragma unroll
        for (int j = 0; j < 2; ++j) {
            int n = n0 + wn + j * 16 + ln;
            if (n >= N) continue;
            float bv = bias ? bias[n] : 0.f;
#pragma unroll
            for (int r = 0; r < 8; ++r) {
                int m = m0 + wm + i * 16 + lg * 8 + r;
                float vlu = acc[i][j][r] * scale + bv;
                long long off = (long long)m * ldc + n;
                if (ADDC) vlu += Cb[off];
                if (GELU) vlu = 0.5f * vlu * (1.f + erff(vlu * 0.70710678118654752f));
                Cb[off] = vlu;
            }
        }
    }
}

// Wt[n*K+k] = (bf16) W[k*N+n]   (weight pre-transpose + convert, once/launch)
__global__ void w_transpose(const float* __restrict__ W, __bf16* __restrict__ Wt,
                            int K, int N)
{
    long long i = (long long)blockIdx.x * blockDim.x + threadIdx.x;
    if (i >= (long long)K * N) return;
    int k = (int)(i % K);
    int n = (int)(i / K);
    Wt[i] = (__bf16)W[(long long)k * N + n];
}

// Row softmax over L elements; one wave per row; output bf16 (feeds P@V WMMA).
__global__ __launch_bounds__(32) void softmax_rows(
    const float* __restrict__ S, __bf16* __restrict__ P, int L)
{
    long long row = blockIdx.x;
    const float* s = S + row * (long long)L;
    __bf16*      p = P + row * (long long)L;
    int lane = threadIdx.x;
    float mx = -1e30f;
    for (int i = lane; i < L; i += 32) mx = fmaxf(mx, s[i]);
#pragma unroll
    for (int o = 16; o; o >>= 1) mx = fmaxf(mx, __shfl_xor(mx, o, 32));
    float sum = 0.f;
    for (int i = lane; i < L; i += 32) sum += __expf(s[i] - mx);
#pragma unroll
    for (int o = 16; o; o >>= 1) sum += __shfl_xor(sum, o, 32);
    float inv = 1.f / sum;
    for (int i = lane; i < L; i += 32) p[i] = (__bf16)(__expf(s[i] - mx) * inv);
}

// x = BN(x [+ t])  (eval-mode batchnorm over feature dim D)
__global__ void add_bn(float* __restrict__ x, const float* __restrict__ t,
                       const float* __restrict__ g, const float* __restrict__ be,
                       const float* __restrict__ mu, const float* __restrict__ var,
                       long long n, int D)
{
    long long i = (long long)blockIdx.x * blockDim.x + threadIdx.x;
    if (i >= n) return;
    int d = (int)(i % D);
    float v = t ? (x[i] + t[i]) : x[i];
    x[i] = (v - mu[d]) * rsqrtf(var[d] + 1e-5f) * g[d] + be[d];
}

// Coarse stage: unshuffle embedded tokens + mask tokens, add pos embedding.
__global__ void build_y(const float* __restrict__ emb, const float* __restrict__ m0,
                        const float* __restrict__ pos, const int* __restrict__ ids_restore,
                        float* __restrict__ y)
{
    long long i = (long long)blockIdx.x * blockDim.x + threadIdx.x;
    if (i >= 64LL * 128 * 256) return;
    int d  = (int)(i & 255);
    int l  = (int)((i >> 8) & 127);
    int bb = (int)(i >> 15);
    int bs = bb >> 3, c = bb & 7;
    int j = ids_restore[(bs * 128 + l) * 8 + c];
    float v = (j < 64) ? emb[((long long)bb * 64 + j) * 256 + d] : m0[d];
    y[i] = v + pos[l * 256 + d];
}

// Fine stage: restore fine tokens + mask tokens.
__global__ void build_fine(const float* __restrict__ xin, const float* __restrict__ m1,
                           const int* __restrict__ ids_restore, float* __restrict__ xf)
{
    long long i = (long long)blockIdx.x * blockDim.x + threadIdx.x;
    if (i >= 64LL * 256 * 256) return;
    int d  = (int)(i & 255);
    int ii = (int)((i >> 8) & 255);
    int bb = (int)(i >> 16);
    int l = ii >> 1, r = ii & 1;
    int bs = bb >> 3, c = bb & 7;
    int j = ids_restore[(bs * 128 + l) * 8 + c] * 2 + r;
    xf[i] = (j < 128) ? xin[((long long)bb * 128 + j) * 256 + d] : m1[d];
}

// cat[row, 0:256]=a[row,:], cat[row, 256:512]=b[row,:]
__global__ void concat2(const float* __restrict__ a, const float* __restrict__ b,
                        float* __restrict__ c, long long rows)
{
    long long i = (long long)blockIdx.x * blockDim.x + threadIdx.x;
    if (i >= rows * 512) return;
    int col = (int)(i & 511);
    long long row = i >> 9;
    c[i] = (col < 256) ? a[row * 256 + col] : b[row * 256 + (col - 256)];
}

// f[(bs*8+c)*256 + l, p] -> out[bs, l, c, p]
__global__ void permute_out(const float* __restrict__ f, float* __restrict__ out)
{
    long long i = (long long)blockIdx.x * blockDim.x + threadIdx.x;
    if (i >= 8LL * 256 * 8 * 16) return;
    int p  = (int)(i & 15);
    int c  = (int)((i >> 4) & 7);
    int l  = (int)((i >> 7) & 255);
    int bs = (int)(i >> 15);
    out[i] = f[(((long long)(bs * 8 + c)) * 256 + l) * 16 + p];
}

// ---------------------------------------------------------------------------

struct Layer {
    const float *qW,*qB,*kW,*kB,*vW,*vB,*oW,*oB;
    const float *g1,*b1,*m1,*v1;
    const float *f1W,*f1B,*f2W,*f2B;
    const float *g2,*b2,*m2,*v2;
};
struct LayerT {   // bf16 [N,K] transposed weights
    const __bf16 *qW,*kW,*vW,*oW,*f1W,*f2W;
};

extern "C" void kernel_launch(void* const* d_in, const int* in_sizes, int n_in,
                              void* d_out, int out_size, void* d_ws, size_t ws_size,
                              hipStream_t stream)
{
    (void)in_sizes; (void)n_in; (void)out_size; (void)ws_size;
    const int B = 64, D = 256, H = 8, Lc = 128, Lf = 256;

    // ---- unpack inputs (setup_inputs insertion order, recursively) ----
    int ii = 0;
    auto nxt = [&]() { return (const float*)d_in[ii++]; };
    const float* x_fine   = nxt();                 // [8,8,128,256]
    const float* x_coarse = nxt();                 // [8,8,64,256]
    const int*   ids      = (const int*)d_in[ii++];// [8,128,8]
    const float *embW = nxt(), *embB = nxt();
    const float *mask0 = nxt(), *mask1 = nxt(), *pos = nxt();
    auto getLayer = [&]() {
        Layer L;
        L.qW=nxt(); L.qB=nxt(); L.kW=nxt(); L.kB=nxt();
        L.vW=nxt(); L.vB=nxt(); L.oW=nxt(); L.oB=nxt();
        L.g1=nxt(); L.b1=nxt(); L.m1=nxt(); L.v1=nxt();
        L.f1W=nxt(); L.f1B=nxt(); L.f2W=nxt(); L.f2B=nxt();
        L.g2=nxt(); L.b2=nxt(); L.m2=nxt(); L.v2=nxt();
        return L;
    };
    Layer enc0[2] = { getLayer(), getLayer() };
    const float *upW = nxt(), *upB = nxt();
    const float *ubG = nxt(), *ubB = nxt(), *ubM = nxt(), *ubV = nxt();
    const float *combW = nxt(), *combB = nxt();
    Layer enc1[2] = { getLayer(), getLayer() };
    const float *finW = nxt(), *finB = nxt();

    // ---- workspace layout ----
    char* wsb = (char*)d_ws;
    size_t off = 0;
    auto alloc = [&](size_t bytes) {
        void* p = wsb + off; off += (bytes + 255) & ~(size_t)255; return p;
    };
    float*  emb    = (float*) alloc((size_t)B * 64 * D * 4);
    float*  x      = (float*) alloc((size_t)B * Lf * D * 4);
    float*  q      = (float*) alloc((size_t)B * Lf * D * 4);
    float*  kbuf   = (float*) alloc((size_t)B * Lf * D * 4);
    float*  vbuf   = (float*) alloc((size_t)B * Lf * D * 4);
    float*  attn   = (float*) alloc((size_t)B * Lf * D * 4);
    float*  tmp1   = (float*) alloc((size_t)B * Lf * 512 * 4);
    float*  tmp2   = (float*) alloc((size_t)B * Lf * D * 4);
    float*  y2     = (float*) alloc((size_t)B * Lf * D * 4);
    float*  xf     = (float*) alloc((size_t)B * Lf * D * 4);
    float*  scores = (float*) alloc((size_t)B * H * Lf * Lf * 4);
    __bf16* probs  = (__bf16*)alloc((size_t)B * H * Lf * Lf * 2);
    float*  fout   = (float*) alloc((size_t)B * Lf * 16 * 4);

    // ---- weight pre-transpose (fp32 [K,N] -> bf16 [N,K]) ----
    auto wt_make = [&](const float* W, int K, int N) -> const __bf16* {
        __bf16* p = (__bf16*)alloc((size_t)K * N * 2);
        long long n = (long long)K * N;
        w_transpose<<<(int)((n + 255) / 256), 256, 0, stream>>>(W, p, K, N);
        return p;
    };
    const __bf16* embWT = wt_make(embW, D, D);
    LayerT enc0T[2], enc1T[2];
    for (int l = 0; l < 2; ++l) {
        const Layer& P = enc0[l];
        enc0T[l] = { wt_make(P.qW, D, D), wt_make(P.kW, D, D), wt_make(P.vW, D, D),
                     wt_make(P.oW, D, D), wt_make(P.f1W, D, 512), wt_make(P.f2W, 512, D) };
    }
    const __bf16* upWT   = wt_make(upW, D, 512);
    const __bf16* combWT = wt_make(combW, 512, D);
    for (int l = 0; l < 2; ++l) {
        const Layer& P = enc1[l];
        enc1T[l] = { wt_make(P.qW, D, D), wt_make(P.kW, D, D), wt_make(P.vW, D, D),
                     wt_make(P.oW, D, D), wt_make(P.f1W, D, 512), wt_make(P.f2W, 512, D) };
    }
    const __bf16* finWT = wt_make(finW, D, 16);

    // ---- launch helpers ----
    auto gemm_lin = [&](const float* A, const __bf16* Wt, const float* bias, float* C,
                        int M, int N, int K, bool gelu) {
        dim3 g((N + 63) / 64, M / 64, 1);
        if (gelu)
            gemm_wmma<float, __bf16, true, true, false><<<g, 128, 0, stream>>>(
                A, 0, 0, K, Wt, 0, 0, K, bias, C, 0, 0, N, M, N, K, 1, 1.f);
        else
            gemm_wmma<float, __bf16, true, false, false><<<g, 128, 0, stream>>>(
                A, 0, 0, K, Wt, 0, 0, K, bias, C, 0, 0, N, M, N, K, 1, 1.f);
    };
    auto scores_l = [&](const float* qb, const float* kb, float* S, int L, bool addprev) {
        dim3 g(L / 64, L / 64, B * H);
        float sc = 0.17677669529663687f;  // 1/sqrt(hd=32)
        if (addprev)
            gemm_wmma<float, float, true, false, true><<<g, 128, 0, stream>>>(
                qb, (long long)L * D, 32, D, kb, (long long)L * D, 32, D, nullptr,
                S, (long long)H * L * L, (long long)L * L, L, L, L, 32, H, sc);
        else
            gemm_wmma<float, float, true, false, false><<<g, 128, 0, stream>>>(
                qb, (long long)L * D, 32, D, kb, (long long)L * D, 32, D, nullptr,
                S, (long long)H * L * L, (long long)L * L, L, L, L, 32, H, sc);
    };
    auto av_l = [&](const __bf16* P, const float* vb, float* ob, int L) {
        dim3 g(1, L / 64, B * H);
        gemm_wmma<__bf16, float, false, false, false><<<g, 128, 0, stream>>>(
            P, (long long)H * L * L, (long long)L * L, L,
            vb, (long long)L * D, 32, D, nullptr,
            ob, (long long)L * D, 32, D, L, 32, L, H, 1.f);
    };
    auto bn_call = [&](float* xb, const float* tb, const Layer& P, bool second, long long n) {
        int blocks = (int)((n + 255) / 256);
        if (second)
            add_bn<<<blocks, 256, 0, stream>>>(xb, tb, P.g2, P.b2, P.m2, P.v2, n, D);
        else
            add_bn<<<blocks, 256, 0, stream>>>(xb, tb, P.g1, P.b1, P.m1, P.v1, n, D);
    };
    auto enc_layer = [&](float* xb, int L, const Layer& P, const LayerT& PT, bool addprev) {
        int M = B * L;
        long long nElem = (long long)M * D;
        gemm_lin(xb, PT.qW, P.qB, q,    M, D, D, false);
        gemm_lin(xb, PT.kW, P.kB, kbuf, M, D, D, false);
        gemm_lin(xb, PT.vW, P.vB, vbuf, M, D, D, false);
        scores_l(q, kbuf, scores, L, addprev);
        softmax_rows<<<B * H * L, 32, 0, stream>>>(scores, probs, L);
        av_l(probs, vbuf, attn, L);
        gemm_lin(attn, PT.oW, P.oB, tmp2, M, D, D, false);
        bn_call(xb, tmp2, P, false, nElem);
        gemm_lin(xb, PT.f1W, P.f1B, tmp1, M, 512, D, true);   // GELU
        gemm_lin(tmp1, PT.f2W, P.f2B, tmp2, M, D, 512, false);
        bn_call(xb, tmp2, P, true, nElem);
    };

    // ---- forward ----
    gemm_lin(x_coarse, embWT, embB, emb, B * 64, D, D, false);
    {
        long long n = 64LL * 128 * 256;
        build_y<<<(int)((n + 255) / 256), 256, 0, stream>>>(emb, mask0, pos, ids, x);
    }
    enc_layer(x, Lc, enc0[0], enc0T[0], false);
    enc_layer(x, Lc, enc0[1], enc0T[1], true);
    gemm_lin(x, upWT, upB, y2, B * Lc, 2 * D, D, false);
    {
        long long n = (long long)B * Lf * D;
        add_bn<<<(int)((n + 255) / 256), 256, 0, stream>>>(
            y2, nullptr, ubG, ubB, ubM, ubV, n, D);
    }
    {
        long long n = 64LL * 256 * 256;
        build_fine<<<(int)((n + 255) / 256), 256, 0, stream>>>(x_fine, mask1, ids, xf);
        long long n2 = 64LL * 256 * 512;
        concat2<<<(int)((n2 + 255) / 256), 256, 0, stream>>>(y2, xf, tmp1, 64LL * 256);
    }
    gemm_lin(tmp1, combWT, combB, x, B * Lf, D, 2 * D, false);
    enc_layer(x, Lf, enc1[0], enc1T[0], false);
    enc_layer(x, Lf, enc1[1], enc1T[1], true);
    gemm_lin(x, finWT, finB, fout, B * Lf, 16, D, false);
    {
        long long n = 8LL * 256 * 8 * 16;
        permute_out<<<(int)((n + 255) / 256), 256, 0, stream>>>(fout, (float*)d_out);
    }
}
